// CausalSelfAttention_3590592660053
// MI455X (gfx1250) — matrix operand order repaired and checked by reading, not compiled, not run on hardware
//
#include <hip/hip_runtime.h>

// ---------------------------------------------------------------------------
// CDNA5 (gfx1250) wave32 WMMA attention block, round 3:
//  - GEMM A-tile staged with GLOBAL_LOAD_ASYNC_TO_LDS_B128 (ASYNCcnt-tracked,
//    no VGPR round trip), published with s_wait_asynccnt + barrier
//  - double-buffered LDS staging in GEMM + attention V path
//  - all matrix math on v_wmma_f32_16x16x32_bf16
// ---------------------------------------------------------------------------

typedef __attribute__((ext_vector_type(16))) __bf16 v16bf;
typedef __attribute__((ext_vector_type(8)))  __bf16 v8bf;
typedef __attribute__((ext_vector_type(8)))  float  v8f;

union AFrag { v16bf v; v8bf h[2]; };

#define WMMA_BF16(A, B, C) \
  __builtin_amdgcn_wmma_f32_16x16x32_bf16(false, (A), false, (B), (short)0, (C), false, false)

// ---- CDNA5 async memory->LDS copy (ISA 10.x / async_tensor §4) ----
// VDST = per-lane LDS byte address, VADDR = per-lane 64-bit global address.
__device__ __forceinline__ void async_load_b128(unsigned lds_off, const void* gaddr) {
  asm volatile("global_load_async_to_lds_b128 %0, %1, off"
               :: "v"(lds_off), "v"(gaddr)
               : "memory");
}
__device__ __forceinline__ void wait_asynccnt0() {
  asm volatile("s_wait_asynccnt 0x0" ::: "memory");
}
// flat shared pointer -> raw LDS byte address (aperture keeps LDS addr in [31:0])
__device__ __forceinline__ unsigned lds_addr(const void* p) {
  return (unsigned)(unsigned long long)p;
}

__device__ __forceinline__ unsigned short f2bfbits(float f) {
  unsigned int u = __float_as_uint(f);
  u += 0x7FFFu + ((u >> 16) & 1u);          // round to nearest even
  return (unsigned short)(u >> 16);
}
__device__ __forceinline__ float bf2f(unsigned short u) {
  return __uint_as_float(((unsigned int)u) << 16);
}

// problem constants
#define TT   2048
#define CC   2048
#define HH   16
#define KVH  4
#define DD   128
#define BTOT 4096          // B*T
#define NQKV 3072          // 2048 q + 512 k + 512 v

// ---------------------------------------------------------------------------
// Kernel 1: fp32 -> bf16 convert with column packing (for wq|wk|wv interleave)
// ---------------------------------------------------------------------------
__global__ void k_convert(const float* __restrict__ src, unsigned short* __restrict__ dst,
                          int n, int ncols, int ldDst, int colOff) {
  int i = blockIdx.x * blockDim.x + threadIdx.x;
  if (i >= n) return;
  int r = i / ncols;
  int c = i - r * ncols;
  dst[(size_t)r * ldDst + colOff + c] = f2bfbits(src[i]);
}

// ---------------------------------------------------------------------------
// Kernel 2: bf16 GEMM, C = A(MxK) * B(KxN), row-major, f32 accum.
// Block tile 128x128, 8 waves of 64(M)x32(N), K-step 32, double-buffered LDS.
// A tile staged asynchronously (memory->LDS DMA), B tile transposed manually.
// ---------------------------------------------------------------------------
__global__ __launch_bounds__(256)
void k_gemm_bf16(const unsigned short* __restrict__ A,
                 const unsigned short* __restrict__ B,
                 float* __restrict__ Cf, unsigned short* __restrict__ Cb,
                 int M, int N, int K, int storeBF) {
  __shared__ unsigned short As[2][128][40];   // [buf][m][k]
  __shared__ unsigned short Bs[2][128][40];   // [buf][n][k] (transposed B tile)

  const int tid  = threadIdx.x;
  const int wid  = tid >> 5;
  const int lane = tid & 31;
  const int lr   = lane & 15;
  const int hi   = (lane >= 16);

  const int bm = blockIdx.y * 128;
  const int bn = blockIdx.x * 128;

  const int wm = (wid & 1) * 64;              // wave grid 2(M) x 4(N)
  const int wn = (wid >> 1) * 32;

  const int ka = hi ? 8 : 0;                  // A frag K-base per lane-half
  const int kb = hi ? 8 : 0;                  // B frag K-base per lane-half

  // cooperative-load coords
  const int ar = tid >> 2;                    // 0..63
  const int ac = (tid & 3) * 8;               // 0,8,16,24
  const int bk = tid >> 4;                    // 0..15
  const int bc = (tid & 15) * 8;              // n offset 0..120

  // A via async DMA to LDS; B loaded to regs and stored transposed
  auto stage = [&](int buf, int k0) {
    async_load_b128(lds_addr(&As[buf][ar][ac]),
                    A + (size_t)(bm + ar)      * K + k0 + ac);
    async_load_b128(lds_addr(&As[buf][ar + 64][ac]),
                    A + (size_t)(bm + ar + 64) * K + k0 + ac);
    const uint4 b0 = *(const uint4*)(B + (size_t)(k0 + bk)      * N + bn + bc);
    const uint4 b1 = *(const uint4*)(B + (size_t)(k0 + bk + 16) * N + bn + bc);
    unsigned short t0[8], t1[8];
    *(uint4*)t0 = b0;
    *(uint4*)t1 = b1;
#pragma unroll
    for (int j = 0; j < 8; ++j) {
      Bs[buf][bc + j][bk]      = t0[j];
      Bs[buf][bc + j][bk + 16] = t1[j];
    }
  };

  v8f acc[4][2];
  const v8f vzero = {0.f,0.f,0.f,0.f,0.f,0.f,0.f,0.f};
#pragma unroll
  for (int mf = 0; mf < 4; ++mf)
#pragma unroll
    for (int nf = 0; nf < 2; ++nf) acc[mf][nf] = vzero;

  stage(0, 0);
  wait_asynccnt0();
  __syncthreads();

  for (int k0 = 0; k0 < K; k0 += 32) {
    const int buf = (k0 >> 5) & 1;
    if (k0 + 32 < K) {
      stage(buf ^ 1, k0 + 32);                // overlap with WMMAs below
      __builtin_prefetch(A + (size_t)(bm + ar) * K + k0 + 64 + ac, 0, 3);
      __builtin_prefetch(B + (size_t)(k0 + 64 + bk) * N + bn + bc, 0, 3);
    }

    AFrag af[4];
#pragma unroll
    for (int mf = 0; mf < 4; ++mf) {
      const unsigned short* p = &As[buf][wm + mf * 16 + lr][0];
      af[mf].h[0] = *(const v8bf*)(p + ka);         // K = ka..ka+7
      af[mf].h[1] = *(const v8bf*)(p + ka + 16);    // K = ka+16..ka+23
    }
    AFrag bfr[2];
#pragma unroll
    for (int nf = 0; nf < 2; ++nf) {
      const unsigned short* p = &Bs[buf][wn + nf * 16 + lr][0];
      bfr[nf].h[0] = *(const v8bf*)(p + kb);        // K = kb..kb+7
      bfr[nf].h[1] = *(const v8bf*)(p + kb + 16);   // K = kb+16..kb+23
    }
#pragma unroll
    for (int mf = 0; mf < 4; ++mf)
#pragma unroll
      for (int nf = 0; nf < 2; ++nf)
        acc[mf][nf] = WMMA_BF16(af[mf].v, bfr[nf].v, acc[mf][nf]);

    wait_asynccnt0();                         // staged tile landed in LDS
    __syncthreads();                          // publish to all waves
  }

  // writeback (C layout: lane = col, vgpr i = row (+8 for hi lane half))
  const int ro = hi ? 8 : 0;
#pragma unroll
  for (int mf = 0; mf < 4; ++mf)
#pragma unroll
    for (int nf = 0; nf < 2; ++nf)
#pragma unroll
      for (int i = 0; i < 8; ++i) {
        const size_t row = bm + wm + mf * 16 + i + ro;
        const size_t col = bn + wn + nf * 16 + lr;
        const float v = acc[mf][nf][i];
        if (storeBF) Cb[row * N + col] = f2bfbits(v);
        else         Cf[row * N + col] = v;
      }
}

// ---------------------------------------------------------------------------
// Kernel 3: RoPE + RMS-norm for q and k heads. One wave per (token, head).
// ---------------------------------------------------------------------------
__global__ __launch_bounds__(32)
void k_rope_rms(const unsigned short* __restrict__ QKV,
                unsigned short* __restrict__ Qo, unsigned short* __restrict__ Ko,
                const float* __restrict__ cosp, const float* __restrict__ sinp) {
  const int row  = blockIdx.x;             // b*T + t
  const int slot = blockIdx.y;             // 0..19
  const int t    = row & (TT - 1);
  const int lane = threadIdx.x;

  const unsigned short* in;
  unsigned short* out;
  if (slot < HH) {
    in  = QKV + (size_t)row * NQKV + slot * DD;
    out = Qo  + (size_t)row * (HH * DD) + slot * DD;
  } else {
    const int j = slot - HH;
    in  = QKV + (size_t)row * NQKV + CC + j * DD;
    out = Ko  + (size_t)row * (KVH * DD) + j * DD;
  }

  float o[4];
  float ss = 0.f;
#pragma unroll
  for (int p = 0; p < 2; ++p) {
    const int d1 = lane + p * 32;          // 0..63
    const float x1 = bf2f(in[d1]);
    const float x2 = bf2f(in[d1 + 64]);
    const float c  = cosp[t * 64 + d1];
    const float s  = sinp[t * 64 + d1];
    const float o1 =  x1 * c + x2 * s;
    const float o2 = -x1 * s + x2 * c;
    o[p * 2]     = o1;
    o[p * 2 + 1] = o2;
    ss += o1 * o1 + o2 * o2;
  }
#pragma unroll
  for (int off = 1; off < 32; off <<= 1) ss += __shfl_xor(ss, off, 32);
  const float r = rsqrtf(ss * (1.0f / 128.0f) + 1.1920928955078125e-07f);
#pragma unroll
  for (int p = 0; p < 2; ++p) {
    out[lane + p * 32]      = f2bfbits(o[p * 2] * r);
    out[lane + p * 32 + 64] = f2bfbits(o[p * 2 + 1] * r);
  }
}

// ---------------------------------------------------------------------------
// Kernel 4: sliding-window flash attention, double-buffered V staging.
// Block = (b, h, 128 q rows); 8 waves, each owns 16 q rows.
// ---------------------------------------------------------------------------
__global__ __launch_bounds__(256)
void k_attn(const unsigned short* __restrict__ Q,     // [BTOT][2048] bf16 (rms+rope)
            const unsigned short* __restrict__ Krot,  // [BTOT][512]  bf16 (rms+rope)
            const unsigned short* __restrict__ QKV,   // V at col 2560+kv*128, ld 3072
            unsigned short* __restrict__ Y,           // [BTOT][2048] bf16
            const int* __restrict__ winp) {
  __shared__ unsigned short Vs[2][128][40];    // [buf][d][key]  (V tile transposed)
  __shared__ unsigned short Ps[8][16][40];     // per-wave P staging [row][key]

  const int tid  = threadIdx.x;
  const int wid  = tid >> 5;
  const int lane = tid & 31;
  const int lr   = lane & 15;
  const int hi   = (lane >= 16);

  const int nQT = TT / 128;
  const int qt  = blockIdx.x % nQT;
  const int h   = (blockIdx.x / nQT) % HH;
  const int b   = blockIdx.x / (nQT * HH);
  const int kv  = h >> 2;
  const int qb  = qt * 128;
  const int win = *winp;

  const int ka = hi ? 8 : 0;                   // A-frag K-base
  const int kb = hi ? 8 : 0;                   // B-frag K-base
  const float scale = 0.08838834764831845f;    // 1/sqrt(128)

  // V cooperative-load coords
  const int vk = tid >> 4;                     // key 0..15 (x2)
  const int vd = (tid & 15) * 8;               // d offset

  auto stageV = [&](int buf, int s0) {
    const uint4 r0 = *(const uint4*)(QKV + (size_t)(b * TT + s0 + vk)      * NQKV
                                     + CC + 512 + kv * DD + vd);
    const uint4 r1 = *(const uint4*)(QKV + (size_t)(b * TT + s0 + vk + 16) * NQKV
                                     + CC + 512 + kv * DD + vd);
    unsigned short t0[8], t1[8];
    *(uint4*)t0 = r0;
    *(uint4*)t1 = r1;
#pragma unroll
    for (int j = 0; j < 8; ++j) {
      Vs[buf][vd + j][vk]      = t0[j];
      Vs[buf][vd + j][vk + 16] = t1[j];
    }
  };

  // ---- preload Q fragments (16x128 row block, direct from global) ----
  AFrag qf[4];
  {
    const unsigned short* qrow =
        Q + (size_t)(b * TT + qb + wid * 16 + lr) * (HH * DD) + h * DD;
#pragma unroll
    for (int kk = 0; kk < 4; ++kk) {
      qf[kk].h[0] = *(const v8bf*)(qrow + kk * 32 + ka);
      qf[kk].h[1] = *(const v8bf*)(qrow + kk * 32 + ka + 16);
    }
  }

  float mrow[8], lrow[8];
  v8f o[8];
  const v8f vzero = {0.f,0.f,0.f,0.f,0.f,0.f,0.f,0.f};
#pragma unroll
  for (int i = 0; i < 8; ++i) { mrow[i] = -3.0e38f; lrow[i] = 0.f; }
#pragma unroll
  for (int df = 0; df < 8; ++df) o[df] = vzero;

  const int s_begin = (qb > win) ? ((qb - win) & ~31) : 0;
  const int s_end   = qb + 128;

  stageV(0, s_begin);
  __syncthreads();

  for (int s0 = s_begin; s0 < s_end; s0 += 32) {
    const int buf = ((s0 - s_begin) >> 5) & 1;
    if (s0 + 32 < s_end) stageV(buf ^ 1, s0 + 32);   // overlap with compute

    // ---- S = Q K^T (K^T fragments straight from global: contiguous in d) ----
    v8f sacc[2] = {vzero, vzero};
    const unsigned short* krow0 =
        Krot + (size_t)(b * TT + s0 + lr) * (KVH * DD) + kv * DD;
    const unsigned short* krow1 =
        Krot + (size_t)(b * TT + s0 + 16 + lr) * (KVH * DD) + kv * DD;
#pragma unroll
    for (int kk = 0; kk < 4; ++kk) {
      AFrag bk0, bk1;
      bk0.h[0] = *(const v8bf*)(krow0 + kk * 32 + kb);
      bk0.h[1] = *(const v8bf*)(krow0 + kk * 32 + kb + 16);
      bk1.h[0] = *(const v8bf*)(krow1 + kk * 32 + kb);
      bk1.h[1] = *(const v8bf*)(krow1 + kk * 32 + kb + 16);
      sacc[0] = WMMA_BF16(qf[kk].v, bk0.v, sacc[0]);
      sacc[1] = WMMA_BF16(qf[kk].v, bk1.v, sacc[1]);
    }

    // ---- online softmax over this 32-key slab ----
    const int rbase = qb + wid * 16 + (hi ? 8 : 0);
    float sv[2][8];
#pragma unroll
    for (int i = 0; i < 8; ++i) {
      const int r = rbase + i;
      float mx = mrow[i];
#pragma unroll
      for (int f = 0; f < 2; ++f) {
        const int c = s0 + f * 16 + lr;
        float x = sacc[f][i] * scale;
        const bool ok = (c <= r) && (r - c <= win);
        x = ok ? x : -3.0e38f;
        sv[f][i] = x;
        mx = fmaxf(mx, x);
      }
#pragma unroll
      for (int off = 1; off < 16; off <<= 1) mx = fmaxf(mx, __shfl_xor(mx, off, 32));

      const float alpha = __expf(mrow[i] - mx);
      mrow[i] = mx;
      lrow[i] *= alpha;
#pragma unroll
      for (int df = 0; df < 8; ++df) o[df][i] *= alpha;

      float psum = 0.f;
#pragma unroll
      for (int f = 0; f < 2; ++f) {
        const float p = (sv[f][i] > -1.0e38f) ? __expf(sv[f][i] - mx) : 0.0f;
        psum += p;
        Ps[wid][i + (hi ? 8 : 0)][f * 16 + lr] = f2bfbits(p);   // [row][key]
      }
#pragma unroll
      for (int off = 1; off < 16; off <<= 1) psum += __shfl_xor(psum, off, 32);
      lrow[i] += psum;
    }

    // ---- P fragment from wave-private LDS (A layout) ----
    AFrag pf;
    {
      const unsigned short* prow = &Ps[wid][lr][0];
      pf.h[0] = *(const v8bf*)(prow + ka);
      pf.h[1] = *(const v8bf*)(prow + ka + 16);
    }

    // ---- O += P V (V fragments contiguous from transposed LDS) ----
#pragma unroll
    for (int df = 0; df < 8; ++df) {
      AFrag bv;
      const unsigned short* vrow = &Vs[buf][df * 16 + lr][0];
      bv.h[0] = *(const v8bf*)(vrow + kb);
      bv.h[1] = *(const v8bf*)(vrow + kb + 16);
      o[df] = WMMA_BF16(pf.v, bv.v, o[df]);
    }
    __syncthreads();                              // release buf for restaging
  }

  // ---- normalize + write y (bf16) ----
#pragma unroll
  for (int i = 0; i < 8; ++i) {
    const float inv = (lrow[i] > 0.f) ? (1.0f / lrow[i]) : 0.0f;
    const size_t row = (size_t)(b * TT + qb + wid * 16 + i + (hi ? 8 : 0));
#pragma unroll
    for (int df = 0; df < 8; ++df)
      Y[row * (HH * DD) + h * DD + df * 16 + lr] = f2bfbits(o[df][i] * inv);
  }
}

// ---------------------------------------------------------------------------
// Launch
// ---------------------------------------------------------------------------
extern "C" void kernel_launch(void* const* d_in, const int* in_sizes, int n_in,
                              void* d_out, int out_size, void* d_ws, size_t ws_size,
                              hipStream_t stream) {
  const float* x    = (const float*)d_in[0];
  const float* cosp = (const float*)d_in[1];
  const float* sinp = (const float*)d_in[2];
  const float* wq   = (const float*)d_in[3];
  const float* wk   = (const float*)d_in[4];
  const float* wv   = (const float*)d_in[5];
  const float* wo   = (const float*)d_in[6];
  const int*   winp = (const int*)d_in[7];
  float* out = (float*)d_out;

  // workspace carve (all bf16 stored as u16)
  unsigned short* ws = (unsigned short*)d_ws;
  unsigned short* xb    = ws;                               // [4096][2048]
  unsigned short* wqkvb = xb    + (size_t)BTOT * CC;        // [2048][3072]
  unsigned short* wob   = wqkvb + (size_t)CC * NQKV;        // [2048][2048]
  unsigned short* qkv   = wob   + (size_t)CC * CC;          // [4096][3072]
  unsigned short* qrot  = qkv   + (size_t)BTOT * NQKV;      // [4096][2048]
  unsigned short* krot  = qrot  + (size_t)BTOT * CC;        // [4096][512]
  unsigned short* ybuf  = krot  + (size_t)BTOT * (KVH*DD);  // [4096][2048]

  // 1) convert fp32 -> bf16 (pack wq|wk|wv into one [2048][3072] matrix)
  {
    int n;
    n = BTOT * CC;
    k_convert<<<(n + 255) / 256, 256, 0, stream>>>(x, xb, n, CC, CC, 0);
    n = CC * CC;
    k_convert<<<(n + 255) / 256, 256, 0, stream>>>(wq, wqkvb, n, CC, NQKV, 0);
    n = CC * (KVH * DD);
    k_convert<<<(n + 255) / 256, 256, 0, stream>>>(wk, wqkvb, n, KVH * DD, NQKV, CC);
    k_convert<<<(n + 255) / 256, 256, 0, stream>>>(wv, wqkvb, n, KVH * DD, NQKV, CC + KVH * DD);
    n = CC * CC;
    k_convert<<<(n + 255) / 256, 256, 0, stream>>>(wo, wob, n, CC, CC, 0);
  }

  // 2) QKV projection: qkv(bf16) = xb @ wqkvb   (M=4096, N=3072, K=2048)
  k_gemm_bf16<<<dim3(NQKV / 128, BTOT / 128), 256, 0, stream>>>(
      xb, wqkvb, nullptr, qkv, BTOT, NQKV, CC, /*storeBF=*/1);

  // 3) RoPE + RMS-norm on q and k heads
  k_rope_rms<<<dim3(BTOT, HH + KVH), 32, 0, stream>>>(qkv, qrot, krot, cosp, sinp);

  // 4) sliding-window attention (512 blocks: 2 batches x 16 heads x 16 q-tiles)
  k_attn<<<dim3(2 * HH * (TT / 128)), 256, 0, stream>>>(qrot, krot, qkv, ybuf, winp);

  // 5) output projection: out(f32) = ybuf @ wob  (M=4096, N=2048, K=2048)
  k_gemm_bf16<<<dim3(CC / 128, BTOT / 128), 256, 0, stream>>>(
      ybuf, wob, out, nullptr, BTOT, CC, CC, /*storeBF=*/0);
}
